// GNNLayer_40931038331316
// MI455X (gfx1250) — compile-verified, matrix operand order checked by reading, not run on hardware
//
#include <hip/hip_runtime.h>
#include <hip/hip_fp16.h>

#define FEAT 128
#define LN_EPS 1e-5f

typedef __attribute__((ext_vector_type(2)))  float    v2f;
typedef __attribute__((ext_vector_type(8)))  float    v8f;
typedef __attribute__((ext_vector_type(16))) _Float16 v16h;

__device__ __forceinline__ void atomic_add_f32(float* p, float v) {
    // native global_atomic_add_f32 (no CAS loop)
    unsafeAtomicAdd(p, v);
}

// ---------------------------------------------------------------- degrees ---
__global__ void k_init_deg(float* __restrict__ deg, int N) {
    int i = blockIdx.x * blockDim.x + threadIdx.x;
    if (i < N) deg[i] = 1.0f;  // self-loop
}

__global__ void k_edge_deg(const long long* __restrict__ dst, float* __restrict__ deg, int E) {
    int e = blockIdx.x * blockDim.x + threadIdx.x;
    if (e < E) atomic_add_f32(&deg[(int)dst[e]], 1.0f);
}

__global__ void k_rsqrt_inplace(float* __restrict__ v, int N) {
    int i = blockIdx.x * blockDim.x + threadIdx.x;
    if (i < N) v[i] = rsqrtf(v[i]);
}

// ------------------------------------------------------------------- GEMM ---
// xw[N,128] = x[N,128] @ W[128,128].
// One 256-thread block (8 wave32) handles 16 rows; wave w computes the 16x16
// output tile at columns [16w, 16w+16) by accumulating 32 WMMA steps of K=4.
#if __has_builtin(__builtin_amdgcn_wmma_f32_16x16x4_f32)
__global__ void k_gemm_xw(const float* __restrict__ x, const float* __restrict__ W,
                          float* __restrict__ xw, int N) {
    const int wave = threadIdx.x >> 5;
    const int lane = threadIdx.x & 31;
    const int row0 = blockIdx.x * 16;
    const int n0   = wave * 16;
    const int m    = lane & 15;            // row (A) / col (B,C) within tile
    const int khalf = (lane >> 4) * 2;     // lanes 0-15 -> K=0,1 ; 16-31 -> K=2,3

    const float* __restrict__ xrow = x + (size_t)(row0 + m) * FEAT;

    v8f c = {};
    #pragma unroll 8
    for (int k0 = 0; k0 < FEAT; k0 += 4) {
        const int k = k0 + khalf;
        v2f a; a.x = xrow[k]; a.y = xrow[k + 1];
        v2f bm;
        bm.x = W[(size_t)k       * FEAT + n0 + m];
        bm.y = W[(size_t)(k + 1) * FEAT + n0 + m];
        c = __builtin_amdgcn_wmma_f32_16x16x4_f32(
                /*neg_a=*/false, a, /*neg_b=*/false, bm,
                /*c_mod=*/(short)0, c, /*reuse_a=*/false, /*reuse_b=*/false);
    }
    // C/D layout: VGPR v -> row v (lanes 0-15) or row v+8 (lanes 16-31), col = lane%16
    const int rbase = row0 + ((lane >> 4) * 8);
    #pragma unroll
    for (int v = 0; v < 8; ++v)
        xw[(size_t)(rbase + v) * FEAT + n0 + m] = c[v];
}
#else
// Fallback: codegen-confirmed f16 WMMA (K=32), converting fp32 operands in-register.
__global__ void k_gemm_xw(const float* __restrict__ x, const float* __restrict__ W,
                          float* __restrict__ xw, int N) {
    const int wave = threadIdx.x >> 5;
    const int lane = threadIdx.x & 31;
    const int row0 = blockIdx.x * 16;
    const int n0   = wave * 16;
    const int m    = lane & 15;
    const int half = lane >> 4;

    const float* __restrict__ xrow = x + (size_t)(row0 + m) * FEAT;

    v8f c = {};
    for (int k0 = 0; k0 < FEAT; k0 += 32) {
        v16h a, bm;
        #pragma unroll
        for (int r = 0; r < 8; ++r) {
            // 16-bit A layout: VGPR r<4 -> K = 2r,2r+1 (+8 for lanes>=16); r>=4 -> +16
            const int k = k0 + ((r & 4) << 2) + 8 * half + 2 * (r & 3);
            a[2 * r]     = (_Float16)xrow[k];
            a[2 * r + 1] = (_Float16)xrow[k + 1];
            bm[2 * r]     = (_Float16)W[(size_t)k       * FEAT + n0 + m];
            bm[2 * r + 1] = (_Float16)W[(size_t)(k + 1) * FEAT + n0 + m];
        }
        c = __builtin_amdgcn_wmma_f32_16x16x32_f16(false, a, false, bm, (short)0, c, false, false);
    }
    const int rbase = row0 + ((lane >> 4) * 8);
    #pragma unroll
    for (int v = 0; v < 8; ++v)
        xw[(size_t)(rbase + v) * FEAT + n0 + m] = c[v];
}
#endif

// -------------------------------------------- self-loop + bias + residual ---
// out = xw*dis^2 (self loop) + b + x  ; seeds the accumulator held in d_out
__global__ void k_base(const float* __restrict__ xw, const float* __restrict__ x,
                       const float* __restrict__ b, const float* __restrict__ dis,
                       float* __restrict__ out, int N) {
    size_t gid = (size_t)blockIdx.x * blockDim.x + threadIdx.x;
    if (gid >= (size_t)N * FEAT) return;
    const int i = (int)(gid >> 7);
    const int d = (int)(gid & (FEAT - 1));
    const float di = dis[i];
    out[gid] = xw[gid] * di * di + x[gid] + b[d];
}

// ------------------------------------------------------------ edge scatter ---
// Thread t: edge e = t>>7, feature d = t&127. One wave = 32 contiguous
// features of one edge (coalesced 128B, scalarized src/dst). xw and out are
// L2-resident (51.2MB each << 192MB), atomics resolve in L2.
__global__ void k_scatter(const long long* __restrict__ edges, const float* __restrict__ xw,
                          const float* __restrict__ dis, float* __restrict__ out, int E) {
    size_t gid = (size_t)blockIdx.x * blockDim.x + threadIdx.x;
    const int e = (int)(gid >> 7);
    const int d = (int)(gid & (FEAT - 1));
    if (e >= E) return;
    const int s = (int)edges[e];
    const int t = (int)edges[(size_t)E + e];
    const float cnorm = dis[s] * dis[t];
    atomic_add_f32(out + (size_t)t * FEAT + d, xw[(size_t)s * FEAT + d] * cnorm);
}

// ------------------------------------------------------- LayerNorm + ReLU ---
// One wave32 per row: 4 floats/lane, shfl_xor tree reduction (wave32-correct).
__global__ void k_lnorm_relu(float* __restrict__ h, const float* __restrict__ gamma,
                             const float* __restrict__ beta, int N) {
    const int wave = threadIdx.x >> 5;
    const int lane = threadIdx.x & 31;
    const int row  = blockIdx.x * 8 + wave;
    if (row >= N) return;

    float4 v = ((const float4*)(h + (size_t)row * FEAT))[lane];
    float s  = v.x + v.y + v.z + v.w;
    float s2 = v.x * v.x + v.y * v.y + v.z * v.z + v.w * v.w;
    #pragma unroll
    for (int off = 16; off >= 1; off >>= 1) {
        s  += __shfl_xor(s,  off, 32);
        s2 += __shfl_xor(s2, off, 32);
    }
    const float mu  = s  * (1.0f / FEAT);
    const float var = s2 * (1.0f / FEAT) - mu * mu;
    const float rs  = rsqrtf(var + LN_EPS);

    const float4 g  = ((const float4*)gamma)[lane];
    const float4 bt = ((const float4*)beta)[lane];
    float4 o;
    o.x = fmaxf((v.x - mu) * rs * g.x + bt.x, 0.0f);
    o.y = fmaxf((v.y - mu) * rs * g.y + bt.y, 0.0f);
    o.z = fmaxf((v.z - mu) * rs * g.z + bt.z, 0.0f);
    o.w = fmaxf((v.w - mu) * rs * g.w + bt.w, 0.0f);
    ((float4*)(h + (size_t)row * FEAT))[lane] = o;
}

// ------------------------------------------------------------------ launch ---
extern "C" void kernel_launch(void* const* d_in, const int* in_sizes, int n_in,
                              void* d_out, int out_size, void* d_ws, size_t ws_size,
                              hipStream_t stream) {
    const float*     x     = (const float*)d_in[0];
    const long long* edges = (const long long*)d_in[1];   // int64 [2, E]
    const float*     W     = (const float*)d_in[2];
    const float*     b     = (const float*)d_in[3];
    const float*     gamma = (const float*)d_in[4];
    const float*     beta  = (const float*)d_in[5];
    float*           out   = (float*)d_out;

    const int N = in_sizes[0] / FEAT;   // 100000
    const int E = in_sizes[1] / 2;      // 1600000

    char*  ws  = (char*)d_ws;
    float* xw  = (float*)ws;                               // N*FEAT floats
    float* dis = (float*)(ws + (size_t)N * FEAT * sizeof(float)); // N floats (deg -> rsqrt)

    const int T = 256;

    // degrees (with self-loop), then dis = deg^{-1/2}, in place
    k_init_deg<<<(N + T - 1) / T, T, 0, stream>>>(dis, N);
    k_edge_deg<<<(E + T - 1) / T, T, 0, stream>>>(edges + E, dis, E);
    k_rsqrt_inplace<<<(N + T - 1) / T, T, 0, stream>>>(dis, N);

    // xw = x @ W  (WMMA; 16 rows per block, 8 wave32 cover 128 cols)
    k_gemm_xw<<<N / 16, 256, 0, stream>>>(x, W, xw, N);

    // seed accumulator in d_out: self-loop + bias + residual
    {
        size_t total = (size_t)N * FEAT;
        k_base<<<(unsigned)((total + T - 1) / T), T, 0, stream>>>(xw, x, b, dis, out, N);
    }

    // scatter-add over edges
    {
        size_t total = (size_t)E * FEAT;
        k_scatter<<<(unsigned)((total + T - 1) / T), T, 0, stream>>>(edges, xw, dis, out, E);
    }

    // LayerNorm + ReLU in place (one wave per row, 8 rows per block)
    k_lnorm_relu<<<(N + 7) / 8, 256, 0, stream>>>(out, gamma, beta, N);
}
